// Model_7335804142279
// MI455X (gfx1250) — matrix operand order repaired and checked by reading, not compiled, not run on hardware
//
#include <hip/hip_runtime.h>
#include <math.h>

// ---- model dims ----
#define BB   4
#define LL   512
#define CIN  21
#define DM   512
#define DI   1024
#define DS   16
#define DTR  32
#define PRED 96
#define ROWS (BB * LL)        // 2048 tokens

typedef __attribute__((ext_vector_type(16))) _Float16 v16h;
typedef __attribute__((ext_vector_type(8)))  _Float16 v8h;
typedef __attribute__((ext_vector_type(8)))  float    v8f;

__device__ __forceinline__ float sigmoidf_(float x) { return 1.0f / (1.0f + __expf(-x)); }

// ---------------- instance norm over time (detached stats) ----------------
__global__ void norm_kernel(const float* __restrict__ x, float* __restrict__ meanv,
                            float* __restrict__ stdv, float* __restrict__ xn) {
  const int b = blockIdx.x / CIN, c = blockIdx.x % CIN;
  __shared__ float s1[128], s2[128];
  float sum = 0.f, sq = 0.f;
  for (int l = threadIdx.x; l < LL; l += 128) {
    float v = x[(b * LL + l) * CIN + c];
    sum += v; sq += v * v;
  }
  s1[threadIdx.x] = sum; s2[threadIdx.x] = sq; __syncthreads();
  for (int o = 64; o > 0; o >>= 1) {
    if (threadIdx.x < o) { s1[threadIdx.x] += s1[threadIdx.x + o]; s2[threadIdx.x] += s2[threadIdx.x + o]; }
    __syncthreads();
  }
  const float m  = s1[0] * (1.0f / LL);
  const float sd = sqrtf(s2[0] * (1.0f / LL) - m * m + 1e-5f);
  if (threadIdx.x == 0) { meanv[b * CIN + c] = m; stdv[b * CIN + c] = sd; }
  const float inv = 1.0f / sd;
  for (int l = threadIdx.x; l < LL; l += 128)
    xn[(b * LL + l) * CIN + c] = (x[(b * LL + l) * CIN + c] - m) * inv;
}

// -------- token embedding: circular conv1d(k=3) + sinusoidal PE -> f16 --------
__global__ void embed_kernel(const float* __restrict__ xn, const float* __restrict__ Wt,
                             _Float16* __restrict__ embH) {
  const int id  = blockIdx.x * blockDim.x + threadIdx.x;   // row*DM + d
  const int row = id >> 9, d = id & (DM - 1);
  const int b = row >> 9, l = row & (LL - 1);
  const float div = __expf(-(float)(d & ~1) * (9.210340371976184f / (float)DM));
  const float ang = (float)l * div;
  float acc = (d & 1) ? __cosf(ang) : __sinf(ang);
  const int lm1 = (l + LL - 1) & (LL - 1);
  const int lp1 = (l + 1) & (LL - 1);
  const float* w  = Wt + d * CIN * 3;
  const float* x0 = xn + (b * LL + lm1) * CIN;
  const float* x1 = xn + (b * LL + l  ) * CIN;
  const float* x2 = xn + (b * LL + lp1) * CIN;
  #pragma unroll
  for (int c = 0; c < CIN; ++c)
    acc += x0[c] * w[c * 3 + 0] + x1[c] * w[c * 3 + 1] + x2[c] * w[c * 3 + 2];
  embH[id] = (_Float16)acc;
}

// ---------------- f32 -> f16 conversion for GEMM weights ----------------
__global__ void cvt_f16(const float* __restrict__ src, _Float16* __restrict__ dst, int n) {
  const int i = blockIdx.x * blockDim.x + threadIdx.x;
  if (i < n) dst[i] = (_Float16)src[i];
}

// ---------------- WMMA GEMM: C[M,N] = A[M,K] * B[N,K]^T (both K-contiguous) ----------------
// Wave tile 32x64: 2 A-frags x 4 shared B-frags -> 8 WMMAs per K-step.
// Explicit ping-pong double buffering over K (K % 64 == 0): loads for the next
// 32-wide K-step issue before the WMMAs of the current one, so s_wait_loadcnt
// retires behind matrix math. All row base pointers hoisted (loop-invariant).
// 8 waves/block stacked in M -> 256x64 block tile (divides N = 64/512/2048).
// 16-bit A 16x32 / B 32x16: lane holds k = 16*(lane/16)+s (contiguous, not ISA table order); f32 C/D per ISA 7.12.2.
struct FragSet {
  v8h a0lo, a0hi, a1lo, a1hi;
  v16h b0, b1, b2, b3;
};

__device__ __forceinline__ void load_frags(FragSet& f, const _Float16* Ar0, const _Float16* Ar1,
                                           const _Float16* Bp0, const _Float16* Bp1,
                                           const _Float16* Bp2, const _Float16* Bp3, int k0) {
  f.a0lo = *(const v8h*)(Ar0 + k0);
  f.a0hi = *(const v8h*)(Ar0 + k0 + 8);
  f.a1lo = *(const v8h*)(Ar1 + k0);
  f.a1hi = *(const v8h*)(Ar1 + k0 + 8);
  f.b0 = *(const v16h*)(Bp0 + k0);
  f.b1 = *(const v16h*)(Bp1 + k0);
  f.b2 = *(const v16h*)(Bp2 + k0);
  f.b3 = *(const v16h*)(Bp3 + k0);
}

__device__ __forceinline__ void wmma_step(const FragSet& f, v8f acc[2][4]) {
  v16h a0, a1;
  #pragma unroll
  for (int e = 0; e < 8; ++e) {
    a0[e] = f.a0lo[e]; a0[e + 8] = f.a0hi[e];
    a1[e] = f.a1lo[e]; a1[e + 8] = f.a1hi[e];
  }
  acc[0][0] = __builtin_amdgcn_wmma_f32_16x16x32_f16(false, a0, false, f.b0, (short)0, acc[0][0], false, false);
  acc[1][0] = __builtin_amdgcn_wmma_f32_16x16x32_f16(false, a1, false, f.b0, (short)0, acc[1][0], false, false);
  acc[0][1] = __builtin_amdgcn_wmma_f32_16x16x32_f16(false, a0, false, f.b1, (short)0, acc[0][1], false, false);
  acc[1][1] = __builtin_amdgcn_wmma_f32_16x16x32_f16(false, a1, false, f.b1, (short)0, acc[1][1], false, false);
  acc[0][2] = __builtin_amdgcn_wmma_f32_16x16x32_f16(false, a0, false, f.b2, (short)0, acc[0][2], false, false);
  acc[1][2] = __builtin_amdgcn_wmma_f32_16x16x32_f16(false, a1, false, f.b2, (short)0, acc[1][2], false, false);
  acc[0][3] = __builtin_amdgcn_wmma_f32_16x16x32_f16(false, a0, false, f.b3, (short)0, acc[0][3], false, false);
  acc[1][3] = __builtin_amdgcn_wmma_f32_16x16x32_f16(false, a1, false, f.b3, (short)0, acc[1][3], false, false);
}

__global__ void wmma_gemm(const _Float16* __restrict__ A, const _Float16* __restrict__ B,
                          float* __restrict__ C, int M, int N, int K) {
  const int lane = threadIdx.x & 31;
  const int wave = threadIdx.x >> 5;
  const int m0 = blockIdx.x * 256 + wave * 32;
  const int n0 = blockIdx.y * 64;
  const int mm = lane & 15;   // A row / B col within 16-tile
  const int hi = lane >> 4;   // K-half selector
  v8f acc[2][4] = {};
  // Loop-invariant row base pointers (lane-dependent K-half offsets folded in).
  const _Float16* Ar0 = A + (size_t)(m0 + mm) * K + hi * 16;
  const _Float16* Ar1 = A + (size_t)(m0 + 16 + mm) * K + hi * 16;
  const _Float16* Bp0 = B + (size_t)(n0 + 0 * 16 + mm) * K + hi * 16;
  const _Float16* Bp1 = B + (size_t)(n0 + 1 * 16 + mm) * K + hi * 16;
  const _Float16* Bp2 = B + (size_t)(n0 + 2 * 16 + mm) * K + hi * 16;
  const _Float16* Bp3 = B + (size_t)(n0 + 3 * 16 + mm) * K + hi * 16;

  FragSet f0, f1;
  load_frags(f0, Ar0, Ar1, Bp0, Bp1, Bp2, Bp3, 0);
  for (int k0 = 0; k0 < K; k0 += 64) {
    load_frags(f1, Ar0, Ar1, Bp0, Bp1, Bp2, Bp3, k0 + 32);   // K%64==0 -> always valid
    wmma_step(f0, acc);
    if (k0 + 64 < K)
      load_frags(f0, Ar0, Ar1, Bp0, Bp1, Bp2, Bp3, k0 + 64);
    wmma_step(f1, acc);
  }
  #pragma unroll
  for (int i = 0; i < 2; ++i) {
    #pragma unroll
    for (int t = 0; t < 4; ++t) {
      #pragma unroll
      for (int r = 0; r < 8; ++r)   // D: row = m0 + i*16 + hi*8 + r, col = n0 + t*16 + mm
        C[(size_t)(m0 + i * 16 + hi * 8 + r) * N + n0 + t * 16 + mm] = acc[i][t][r];
    }
  }
}

// ---------------- depthwise causal conv(k=4) + SiLU ----------------
__global__ void conv_silu_kernel(const float* __restrict__ xz, const float* __restrict__ cw,
                                 const float* __restrict__ cb, float* __restrict__ xc,
                                 _Float16* __restrict__ xcH) {
  const int id  = blockIdx.x * blockDim.x + threadIdx.x;   // row*DI + d
  const int row = id >> 10, d = id & (DI - 1);
  const int b = row >> 9, l = row & (LL - 1);
  float acc = cb[d];
  #pragma unroll
  for (int j = 0; j < 4; ++j) {
    const int ll = l - 3 + j;
    if (ll >= 0) acc += xz[(size_t)(b * LL + ll) * (2 * DI) + d] * cw[d * 4 + j];
  }
  const float s = acc * sigmoidf_(acc);
  xc[id] = s;
  xcH[id] = (_Float16)s;
}

// ---------------- dt = softplus(dtp @ W_dt^T + b_dt) ----------------
__global__ void dt_kernel(const float* __restrict__ dbl, const float* __restrict__ Wdt,
                          const float* __restrict__ bdt, float* __restrict__ dt) {
  const int id  = blockIdx.x * blockDim.x + threadIdx.x;   // row*DI + d
  const int row = id >> 10, d = id & (DI - 1);
  float acc = bdt[d];
  const float* p = dbl + (size_t)row * 64;      // cols 0..31 = dtp
  const float* w = Wdt + d * DTR;
  #pragma unroll
  for (int r = 0; r < DTR; ++r) acc += p[r] * w[r];
  dt[id] = (acc > 20.f) ? acc : __logf(1.f + __expf(acc));
}

// -------- fused selective scan, 4-way state split --------
// 4 adjacent lanes share one (b, d) recurrence; each owns 4 of the 16 states.
// dA/dBx computed on the fly (never materialized); per-step y summed with 2x shfl_xor;
// output gated with silu(z) and written as f16 GEMM input.
__global__ void scan_kernel(const float* __restrict__ dt, const float* __restrict__ xc,
                            const float* __restrict__ dbl, const float* __restrict__ xz,
                            const float* __restrict__ A_log, const float* __restrict__ Dp,
                            _Float16* __restrict__ ygH) {
  const int tid   = blockIdx.x * blockDim.x + threadIdx.x;   // 0 .. BB*DI*4-1
  const int group = tid >> 2;                                 // (b, d)
  const int q     = tid & 3;                                  // state quarter
  const int b = group >> 10, d = group & (DI - 1);
  const int sb = q * 4;
  float Aneg[4], h[4];
  #pragma unroll
  for (int s = 0; s < 4; ++s) { Aneg[s] = -__expf(A_log[d * DS + sb + s]); h[s] = 0.f; }
  const float Dd = Dp[d];
  for (int l = 0; l < LL; ++l) {
    const int row = b * LL + l;
    const float dtv = dt[(size_t)row * DI + d];
    const float xv  = xc[(size_t)row * DI + d];
    const float zv  = xz[(size_t)row * (2 * DI) + DI + d];
    const float4 Bm = *(const float4*)(dbl + (size_t)row * 64 + 32 + sb);
    const float4 Cm = *(const float4*)(dbl + (size_t)row * 64 + 48 + sb);
    const float dtx = dtv * xv;
    h[0] = __expf(dtv * Aneg[0]) * h[0] + dtx * Bm.x;
    h[1] = __expf(dtv * Aneg[1]) * h[1] + dtx * Bm.y;
    h[2] = __expf(dtv * Aneg[2]) * h[2] + dtx * Bm.z;
    h[3] = __expf(dtv * Aneg[3]) * h[3] + dtx * Bm.w;
    float y = h[0] * Cm.x + h[1] * Cm.y + h[2] * Cm.z + h[3] * Cm.w;
    y += __shfl_xor(y, 1, 32);
    y += __shfl_xor(y, 2, 32);
    if (q == 0) {
      const float yg = (y + xv * Dd) * (zv * sigmoidf_(zv));
      ygH[(size_t)row * DI + d] = (_Float16)yg;
    }
  }
}

// ---------------- p = mo @ W_proj^T + b_proj  (N=21, small) ----------------
__global__ void proj_kernel(const float* __restrict__ mo, const float* __restrict__ Wp,
                            const float* __restrict__ bp, float* __restrict__ p) {
  const int id = blockIdx.x * blockDim.x + threadIdx.x;
  if (id >= ROWS * CIN) return;
  const int row = id / CIN, c = id % CIN;
  float acc = bp[c];
  const float* m = mo + (size_t)row * DM;
  const float* w = Wp + c * DM;
  for (int d = 0; d < DM; ++d) acc += m[d] * w[d];
  p[id] = acc;
}

// ---------------- temporal mixing (L->PRED) + de-normalization ----------------
__global__ void out_kernel(const float* __restrict__ p, const float* __restrict__ Wol,
                           const float* __restrict__ bol, const float* __restrict__ meanv,
                           const float* __restrict__ stdv, float* __restrict__ out) {
  const int id = blockIdx.x * blockDim.x + threadIdx.x;
  if (id >= BB * PRED * CIN) return;
  const int b = id / (PRED * CIN);
  const int rem = id % (PRED * CIN);
  const int pi = rem / CIN, c = rem % CIN;
  float acc = bol[pi];
  const float* w  = Wol + pi * LL;
  const float* pc = p + (size_t)b * LL * CIN + c;
  for (int l = 0; l < LL; ++l) acc += pc[(size_t)l * CIN] * w[l];
  out[id] = acc * stdv[b * CIN + c] + meanv[b * CIN + c];
}

extern "C" void kernel_launch(void* const* d_in, const int* in_sizes, int n_in,
                              void* d_out, int out_size, void* d_ws, size_t ws_size,
                              hipStream_t stream) {
  (void)in_sizes; (void)n_in; (void)out_size; (void)ws_size;
  const float* x_enc   = (const float*)d_in[0];
  const float* W_token = (const float*)d_in[1];
  const float* W_in    = (const float*)d_in[2];
  const float* conv_w  = (const float*)d_in[3];
  const float* conv_b  = (const float*)d_in[4];
  const float* W_xproj = (const float*)d_in[5];
  const float* W_dt    = (const float*)d_in[6];
  const float* b_dt    = (const float*)d_in[7];
  const float* A_log   = (const float*)d_in[8];
  const float* Dp      = (const float*)d_in[9];
  const float* W_outp  = (const float*)d_in[10];
  const float* W_proj  = (const float*)d_in[11];
  const float* b_proj  = (const float*)d_in[12];
  const float* W_ol    = (const float*)d_in[13];
  const float* b_ol    = (const float*)d_in[14];
  float* out = (float*)d_out;

  char* base = (char*)d_ws;
  size_t off = 0;
  auto alloc = [&](size_t bytes) -> char* {
    char* p = base + off;
    off += (bytes + 255) & ~(size_t)255;
    return p;
  };
  float*    meanv = (float*)   alloc((size_t)BB * CIN * 4);
  float*    stdv  = (float*)   alloc((size_t)BB * CIN * 4);
  float*    xn    = (float*)   alloc((size_t)BB * LL * CIN * 4);
  _Float16* embH  = (_Float16*)alloc((size_t)ROWS * DM * 2);
  _Float16* WinH  = (_Float16*)alloc((size_t)(2 * DI) * DM * 2);
  _Float16* WxH   = (_Float16*)alloc((size_t)(DTR + 2 * DS) * DI * 2);
  _Float16* WoH   = (_Float16*)alloc((size_t)DM * DI * 2);
  float*    xz    = (float*)   alloc((size_t)ROWS * (2 * DI) * 4);
  float*    xc    = (float*)   alloc((size_t)ROWS * DI * 4);
  _Float16* xcH   = (_Float16*)alloc((size_t)ROWS * DI * 2);
  float*    dbl   = (float*)   alloc((size_t)ROWS * 64 * 4);
  float*    dtb   = (float*)   alloc((size_t)ROWS * DI * 4);
  _Float16* ygH   = (_Float16*)alloc((size_t)ROWS * DI * 2);
  float*    mo    = (float*)   alloc((size_t)ROWS * DM * 4);
  float*    pbuf  = (float*)   alloc((size_t)ROWS * CIN * 4);

  norm_kernel<<<BB * CIN, 128, 0, stream>>>(x_enc, meanv, stdv, xn);
  embed_kernel<<<(ROWS * DM) / 256, 256, 0, stream>>>(xn, W_token, embH);
  cvt_f16<<<((2 * DI) * DM) / 256, 256, 0, stream>>>(W_in, WinH, (2 * DI) * DM);
  cvt_f16<<<((DTR + 2 * DS) * DI) / 256, 256, 0, stream>>>(W_xproj, WxH, (DTR + 2 * DS) * DI);
  cvt_f16<<<(DM * DI) / 256, 256, 0, stream>>>(W_outp, WoH, DM * DI);

  // in_proj: [2048 x 2048] = embH[2048 x 512] * W_in^T
  wmma_gemm<<<dim3(ROWS / 256, (2 * DI) / 64), 256, 0, stream>>>(embH, WinH, xz, ROWS, 2 * DI, DM);
  conv_silu_kernel<<<(ROWS * DI) / 256, 256, 0, stream>>>(xz, conv_w, conv_b, xc, xcH);
  // x_proj: [2048 x 64] = xconv[2048 x 1024] * W_xproj^T
  wmma_gemm<<<dim3(ROWS / 256, 1), 256, 0, stream>>>(xcH, WxH, dbl, ROWS, 64, DI);
  dt_kernel<<<(ROWS * DI) / 256, 256, 0, stream>>>(dbl, W_dt, b_dt, dtb);
  // scan: 4 lanes per (b,d), 4 states each -> 16384 threads / 512 waves
  scan_kernel<<<(BB * DI * 4) / 256, 256, 0, stream>>>(dtb, xc, dbl, xz, A_log, Dp, ygH);
  // out_proj: [2048 x 512] = yg[2048 x 1024] * W_outp^T
  wmma_gemm<<<dim3(ROWS / 256, DM / 64), 256, 0, stream>>>(ygH, WoH, mo, ROWS, DM, DI);
  proj_kernel<<<(ROWS * CIN + 255) / 256, 256, 0, stream>>>(mo, W_proj, b_proj, pbuf);
  out_kernel<<<(BB * PRED * CIN + 255) / 256, 256, 0, stream>>>(pbuf, W_ol, b_ol, meanv, stdv, out);
}
